// GConvLSTMCellspatial_89515708383432
// MI455X (gfx1250) — matrix operand order repaired and checked by reading, not compiled, not run on hardware
//
#include <hip/hip_runtime.h>
#include <hip/hip_bf16.h>
#include <stdint.h>

#define CIN   128
#define HDIM  128
#define WCOLS 512   // 4 gates * 128

typedef __attribute__((ext_vector_type(16))) __bf16       v16bf;
typedef __attribute__((ext_vector_type(8)))  float        v8f;
typedef __attribute__((ext_vector_type(4)))  unsigned int v4u;
typedef __attribute__((ext_vector_type(4)))  float        v4f;

union V16BF { v16bf v; v4u q[2]; };

__device__ __forceinline__ unsigned short f2bf(float f) {
  unsigned int u = __float_as_uint(f);
  u += 0x7FFFu + ((u >> 16) & 1u);          // round-to-nearest-even
  return (unsigned short)(u >> 16);
}

__device__ __forceinline__ float sigmoidf(float x) {
  return 1.0f / (1.0f + __expf(-x));
}

// ---------------- degree / norm ----------------
__global__ void k_deg_init(float* deg, int N) {
  int i = blockIdx.x * blockDim.x + threadIdx.x;
  if (i < N) deg[i] = 1.0f;                 // self-loop weight
}

__global__ void k_deg_acc(const int* __restrict__ dst, const float* __restrict__ w,
                          float* deg, int E, int N) {
  int e = blockIdx.x * blockDim.x + threadIdx.x;
  if (e >= E) return;
  int d = dst[e];
  if ((unsigned)d < (unsigned)N) atomicAdd(&deg[d], w[e]);
}

__global__ void k_deg_inv(const float* __restrict__ deg, float* dis, int N) {
  int i = blockIdx.x * blockDim.x + threadIdx.x;
  if (i >= N) return;
  float d = deg[i];
  dis[i] = (d > 0.0f) ? rsqrtf(fmaxf(d, 1e-12f)) : 0.0f;
}

__global__ void k_edge_norm(const int* __restrict__ src, const int* __restrict__ dst,
                            const float* __restrict__ w, const float* __restrict__ dis,
                            float* nrm, int E, int N) {
  int e = blockIdx.x * blockDim.x + threadIdx.x;
  if (e >= E) return;
  int s = src[e], d = dst[e];
  float ds = ((unsigned)s < (unsigned)N) ? dis[s] : 0.0f;
  float dd = ((unsigned)d < (unsigned)N) ? dis[d] : 0.0f;
  nrm[e] = ds * w[e] * dd;
}

// ---------------- GCN propagation: out = D^-1/2 (A+I) D^-1/2 in ----------------
// self-loop term: out[i] = dis[i]^2 * in[i]  (also zero-initializes out)
__global__ void k_prop_self(const float* __restrict__ in, float* __restrict__ out,
                            const float* __restrict__ dis, int N, int W) {
  long long i = (long long)blockIdx.x * blockDim.x + threadIdx.x;
  int nvec = W >> 2;
  long long total = (long long)N * nvec;
  if (i >= total) return;
  int node = (int)(i / nvec);
  float s = dis[node]; s = s * s;
  v4f x = ((const v4f*)in)[i];
  ((v4f*)out)[i] = x * s;
}

// one wave per edge; lane covers W/128 float4 chunks; f32 atomics land in L2
__global__ void k_prop_edge(const float* __restrict__ in, float* __restrict__ out,
                            const int* __restrict__ src, const int* __restrict__ dst,
                            const float* __restrict__ nrm, int E, int N, int W) {
  long long gid = (long long)blockIdx.x * blockDim.x + threadIdx.x;
  int e = (int)(gid >> 5);
  int lane = (int)(gid & 31);
  if (e >= E) return;
  int s = src[e], d = dst[e];
  if ((unsigned)s >= (unsigned)N || (unsigned)d >= (unsigned)N) return;
  float w = nrm[e];
  const v4f* ip = (const v4f*)(in + (size_t)s * W);
  float* op = out + (size_t)d * W;
  // warm the destination cachelines before the atomic burst (global_prefetch_b8)
  __builtin_prefetch(op + (lane << 2), 0, 1);
  int nv = W >> 7;                          // float4s per lane
#pragma unroll
  for (int v = 0; v < nv; ++v) {
    int c = lane + (v << 5);
    v4f x = ip[c] * w;
    atomicAdd(op + (c << 2) + 0, x[0]);
    atomicAdd(op + (c << 2) + 1, x[1]);
    atomicAdd(op + (c << 2) + 2, x[2]);
    atomicAdd(op + (c << 2) + 3, x[3]);
  }
}

// ---------------- f32 -> bf16, 8 elements/thread (b128 loads + b128 store) ------
__global__ void k_cvt_bf16x8(const float* __restrict__ in,
                             unsigned short* __restrict__ out, long long count8) {
  long long i = (long long)blockIdx.x * blockDim.x + threadIdx.x;
  if (i >= count8) return;
  const v4f* ip = (const v4f*)in + (i << 1);
  v4f a = ip[0], b = ip[1];
  v4u o;
  o[0] = (unsigned)f2bf(a[0]) | ((unsigned)f2bf(a[1]) << 16);
  o[1] = (unsigned)f2bf(a[2]) | ((unsigned)f2bf(a[3]) << 16);
  o[2] = (unsigned)f2bf(b[0]) | ((unsigned)f2bf(b[1]) << 16);
  o[3] = (unsigned)f2bf(b[2]) | ((unsigned)f2bf(b[3]) << 16);
  ((v4u*)out)[i] = o;
}

// pack W (flat: g*16384 + k*128 + n_in_gate, f32) into the per-lane WMMA B layout:
// packed[((ct*4+ks)*32+lane)*16 + j] = W[k = ks*32 + (lane>=16)*16 + j][n = ct*16 + lane%16]
__global__ void k_pack_w(const float* __restrict__ w, unsigned short* __restrict__ out) {
  int tid = blockIdx.x * blockDim.x + threadIdx.x;
  if (tid >= 32 * 4 * 32 * 16) return;
  int j    = tid & 15;
  int lane = (tid >> 4) & 31;
  int ks   = (tid >> 9) & 3;
  int ct   = tid >> 11;
  int n = (ct << 4) + (lane & 15);          // global output column (0..511)
  int k = (ks << 5) + ((lane >> 4) << 4) + j;
  out[tid] = f2bf(w[((n >> 7) << 14) + (k << 7) + (n & 127)]);
}

// ---------------- WMMA GEMM: C[M,512] = act(A[M,128(bf16)] @ Wp + bias) ----------------
// one 16x16 tile per wave, K=128 as 4x v_wmma_f32_16x16x32_bf16
__global__ void k_gemm_wmma(const unsigned short* __restrict__ A, int lda,
                            const unsigned short* __restrict__ Wp,
                            const float* __restrict__ bias,
                            float* __restrict__ C, int ldc,
                            int N, int rowTiles, int colTiles,
                            int blockDiag, int relu, int accum) {
  int wv = threadIdx.x >> 5;
  int lane = threadIdx.x & 31;
  int t = blockIdx.x * 8 + wv;
  if (t >= rowTiles * colTiles) return;
  int mt = t / colTiles, ct = t % colTiles;
  int m0 = mt << 4;
  int row = m0 + (lane & 15);
  if (row >= N) row = N - 1;
  int aColOff = blockDiag ? ((ct >> 3) << 7) : 0;   // per-gate K-offset for GEMM2
  int off = (lane < 16) ? 0 : 16;                   // contiguous K layout (matches packed B)
  const unsigned short* Abase = A + (size_t)row * lda + aColOff;
  const unsigned short* Wbase = Wp + ((((size_t)ct << 2) << 5) + lane) * 16;

  v8f acc = {0.f, 0.f, 0.f, 0.f, 0.f, 0.f, 0.f, 0.f};
#pragma unroll
  for (int ks = 0; ks < 4; ++ks) {
    V16BF a, b;
    a.q[0] = *(const v4u*)(Abase + ks * 32 + off);        // K = ks*32 + off .. +7
    a.q[1] = *(const v4u*)(Abase + ks * 32 + off + 8);    // K = ks*32 + off+8 .. +7
    const v4u* wp4 = (const v4u*)(Wbase + (size_t)ks * 512);
    b.q[0] = wp4[0];
    b.q[1] = wp4[1];
    acc = __builtin_amdgcn_wmma_f32_16x16x32_bf16(
        false, a.v, false, b.v, (short)0, acc, false, false);
  }

  int n = (ct << 4) + (lane & 15);
  float bval = bias ? bias[n] : 0.0f;
  int mrow0 = m0 + ((lane < 16) ? 0 : 8);
#pragma unroll
  for (int r = 0; r < 8; ++r) {
    int m = mrow0 + r;
    if (m >= N) continue;
    float v = acc[r] + bval;
    if (relu) v = fmaxf(v, 0.0f);
    size_t idx = (size_t)m * ldc + n;
    if (accum) v += C[idx];
    C[idx] = v;
  }
}

// ---------------- peephole LSTM pointwise ----------------
__global__ void k_lstm(const float* __restrict__ G, const float* __restrict__ c,
                       const float* __restrict__ wc, const float* __restrict__ bg,
                       float* __restrict__ out, int N) {
  int gid = blockIdx.x * blockDim.x + threadIdx.x;
  if (gid >= N * HDIM) return;
  int node = gid >> 7;
  int hh = gid & 127;
  const float* g = G + (size_t)node * WCOLS;
  float gi = g[hh], gf = g[HDIM + hh], gc = g[2 * HDIM + hh], go = g[3 * HDIM + hh];
  float cv = c[gid];
  float I = sigmoidf(gi + wc[hh] * cv + bg[hh]);
  float F = sigmoidf(gf + wc[HDIM + hh] * cv + bg[HDIM + hh]);
  float T = tanhf(gc + bg[2 * HDIM + hh]);
  float Cn = F * cv + I * T;
  float O = sigmoidf(go + wc[2 * HDIM + hh] * Cn + bg[3 * HDIM + hh]);
  float Hn = O * tanhf(Cn);
  out[gid] = Hn;
  out[(size_t)N * HDIM + gid] = Cn;
}

extern "C" void kernel_launch(void* const* d_in, const int* in_sizes, int n_in,
                              void* d_out, int out_size, void* d_ws, size_t ws_size,
                              hipStream_t stream) {
  const float* x   = (const float*)d_in[0];
  const int*   ei  = (const int*)d_in[1];
  const float* ew  = (const float*)d_in[2];
  const float* h   = (const float*)d_in[3];
  const float* c   = (const float*)d_in[4];
  const float* Wx0 = (const float*)d_in[5];
  const float* bx0 = (const float*)d_in[6];
  const float* Wx1 = (const float*)d_in[7];
  const float* bx1 = (const float*)d_in[8];
  const float* Wh0 = (const float*)d_in[9];
  const float* bh0 = (const float*)d_in[10];
  const float* Wh1 = (const float*)d_in[11];
  const float* bh1 = (const float*)d_in[12];
  const float* wc  = (const float*)d_in[13];
  const float* bg  = (const float*)d_in[14];

  const int N = in_sizes[0] / CIN;
  const int E = in_sizes[2];
  const int* src = ei;
  const int* dst = ei + E;

  // ---- workspace carve (256B aligned) ----
  char* ws = (char*)d_ws;
  size_t off = 0;
  auto alloc = [&](size_t bytes) -> void* {
    off = (off + 255) & ~(size_t)255;
    void* p = ws + off;
    off += bytes;
    return p;
  };
  float* deg  = (float*)alloc((size_t)N * 4);
  float* dis  = (float*)alloc((size_t)N * 4);
  float* nrm  = (float*)alloc((size_t)E * 4);
  float* ax   = (float*)alloc((size_t)N * CIN * 4);
  float* ah   = (float*)alloc((size_t)N * CIN * 4);
  unsigned short* axb = (unsigned short*)alloc((size_t)N * CIN * 2);
  unsigned short* ahb = (unsigned short*)alloc((size_t)N * CIN * 2);
  float* Y0x  = (float*)alloc((size_t)N * WCOLS * 4);
  float* Y0h  = (float*)alloc((size_t)N * WCOLS * 4);
  float* a1x  = (float*)alloc((size_t)N * WCOLS * 4);
  float* a1h  = (float*)alloc((size_t)N * WCOLS * 4);
  unsigned short* a1xb = (unsigned short*)alloc((size_t)N * WCOLS * 2);
  unsigned short* a1hb = (unsigned short*)alloc((size_t)N * WCOLS * 2);
  unsigned short* Wx0p = (unsigned short*)alloc(65536 * 2);
  unsigned short* Wh0p = (unsigned short*)alloc(65536 * 2);
  unsigned short* Wx1p = (unsigned short*)alloc(65536 * 2);
  unsigned short* Wh1p = (unsigned short*)alloc(65536 * 2);
  float* G = Y0x;  // Y0x is dead after its prop; reuse for gate pre-activations

  const int T = 256;
  auto blocks = [](long long n, int t) { return (unsigned)((n + t - 1) / t); };

  // 1) degree / norm
  k_deg_init<<<blocks(N, T), T, 0, stream>>>(deg, N);
  k_deg_acc<<<blocks(E, T), T, 0, stream>>>(dst, ew, deg, E, N);
  k_deg_inv<<<blocks(N, T), T, 0, stream>>>(deg, dis, N);
  k_edge_norm<<<blocks(E, T), T, 0, stream>>>(src, dst, ew, dis, nrm, E, N);

  // 2) pack weights into WMMA B-lane layout (bf16)
  k_pack_w<<<256, T, 0, stream>>>(Wx0, Wx0p);
  k_pack_w<<<256, T, 0, stream>>>(Wh0, Wh0p);
  k_pack_w<<<256, T, 0, stream>>>(Wx1, Wx1p);
  k_pack_w<<<256, T, 0, stream>>>(Wh1, Wh1p);

  // 3) first propagation (width 128) on x and h
  k_prop_self<<<blocks((long long)N * 32, T), T, 0, stream>>>(x, ax, dis, N, CIN);
  k_prop_edge<<<blocks((long long)E * 32, T), T, 0, stream>>>(x, ax, src, dst, nrm, E, N, CIN);
  k_prop_self<<<blocks((long long)N * 32, T), T, 0, stream>>>(h, ah, dis, N, CIN);
  k_prop_edge<<<blocks((long long)E * 32, T), T, 0, stream>>>(h, ah, src, dst, nrm, E, N, CIN);
  k_cvt_bf16x8<<<blocks((long long)N * CIN / 8, T), T, 0, stream>>>(ax, axb, (long long)N * CIN / 8);
  k_cvt_bf16x8<<<blocks((long long)N * CIN / 8, T), T, 0, stream>>>(ah, ahb, (long long)N * CIN / 8);

  // 4) GEMM1: Y0 = relu(agg @ W0 + b0), all 4 gates as 512 cols
  const int rowTiles = (N + 15) / 16, colTiles = WCOLS / 16;
  const unsigned gblk = blocks((long long)rowTiles * colTiles, 8);
  k_gemm_wmma<<<gblk, T, 0, stream>>>(axb, CIN, Wx0p, bx0, Y0x, WCOLS,
                                      N, rowTiles, colTiles, 0, 1, 0);
  k_gemm_wmma<<<gblk, T, 0, stream>>>(ahb, CIN, Wh0p, bh0, Y0h, WCOLS,
                                      N, rowTiles, colTiles, 0, 1, 0);

  // 5) second propagation (width 512)
  k_prop_self<<<blocks((long long)N * 128, T), T, 0, stream>>>(Y0x, a1x, dis, N, WCOLS);
  k_prop_edge<<<blocks((long long)E * 32, T), T, 0, stream>>>(Y0x, a1x, src, dst, nrm, E, N, WCOLS);
  k_prop_self<<<blocks((long long)N * 128, T), T, 0, stream>>>(Y0h, a1h, dis, N, WCOLS);
  k_prop_edge<<<blocks((long long)E * 32, T), T, 0, stream>>>(Y0h, a1h, src, dst, nrm, E, N, WCOLS);
  k_cvt_bf16x8<<<blocks((long long)N * WCOLS / 8, T), T, 0, stream>>>(a1x, a1xb, (long long)N * WCOLS / 8);
  k_cvt_bf16x8<<<blocks((long long)N * WCOLS / 8, T), T, 0, stream>>>(a1h, a1hb, (long long)N * WCOLS / 8);

  // 6) GEMM2 (block-diagonal per gate): G = a1x@Wx1 + bx1, then G += a1h@Wh1 + bh1
  k_gemm_wmma<<<gblk, T, 0, stream>>>(a1xb, WCOLS, Wx1p, bx1, G, WCOLS,
                                      N, rowTiles, colTiles, 1, 0, 0);
  k_gemm_wmma<<<gblk, T, 0, stream>>>(a1hb, WCOLS, Wh1p, bh1, G, WCOLS,
                                      N, rowTiles, colTiles, 1, 0, 1);

  // 7) LSTM pointwise -> (Hn, Cn)
  k_lstm<<<blocks((long long)N * HDIM, T), T, 0, stream>>>(G, c, wc, bg, (float*)d_out, N);
}
